// MoELayer_10642928959648
// MI455X (gfx1250) — matrix operand order repaired and checked by reading, not compiled, not run on hardware
//
#include <hip/hip_runtime.h>
#include <math.h>

// ---------------- problem constants ----------------
#define T_TOK 8192   // B*S tokens
#define HDIM  1024
#define FDIM  4096
#define NEXP  8
#define CAP   4096   // per-expert capacity (expected ~2048)
#define SA    40     // LDS tile row stride in bf16 elems (32 data + 8 pad) -> conflict-free b128

typedef __attribute__((ext_vector_type(8)))  __bf16 v8bf;
typedef __attribute__((ext_vector_type(16))) __bf16 v16bf;
typedef __attribute__((ext_vector_type(8)))  float  v8f;
typedef int v4i_vs __attribute__((vector_size(16)));
typedef unsigned short ushort_t;

static __device__ __forceinline__ ushort_t f2bf(float f) {
  return __builtin_bit_cast(ushort_t, (__bf16)f);
}

static __device__ __forceinline__ v16bf cat8(v8bf lo, v8bf hi) {
  return __builtin_shufflevector(lo, hi, 0,1,2,3,4,5,6,7,8,9,10,11,12,13,14,15);
}

// A fragment (16x32 bf16): lane m = lane&15; half = lane>>4 selects K chunks
// [half*8, +8) and [16+half*8, +8): two contiguous 16B LDS reads.
static __device__ __forceinline__ v16bf load_fragA(const ushort_t* lds, int mRow, int half) {
  const v8bf lo = *reinterpret_cast<const v8bf*>(lds + mRow * SA + half * 8);
  const v8bf hi = *reinterpret_cast<const v8bf*>(lds + mRow * SA + 16 + half * 8);
  return cat8(lo, hi);
}

// B fragment (32x16 bf16): lane n = lane&15; half selects K chunks
// [half*8, +8) and [16+half*8, +8) of column n (tile stored [n][k]).
static __device__ __forceinline__ v16bf load_fragB(const ushort_t* lds, int nCol, int half) {
  const v8bf lo = *reinterpret_cast<const v8bf*>(lds + nCol * SA + half * 8);
  const v8bf hi = *reinterpret_cast<const v8bf*>(lds + nCol * SA + 16 + half * 8);
  return cat8(lo, hi);
}

// ---------------- async global->LDS staging (CDNA5), with portable fallback ----
#if __has_builtin(__builtin_amdgcn_global_load_async_to_lds_b128)
#define ASYNC_LDS 1
#endif

static __device__ __forceinline__ void copy16(const ushort_t* g, ushort_t* l) {
#ifdef ASYNC_LDS
  __builtin_amdgcn_global_load_async_to_lds_b128(
      (__attribute__((address_space(1))) v4i_vs*)g,
      (__attribute__((address_space(3))) v4i_vs*)l, 0, 0);
#else
  *reinterpret_cast<uint4*>(l) = *reinterpret_cast<const uint4*>(g);
#endif
}

static __device__ __forceinline__ void lds_stage_fence() {
#ifdef ASYNC_LDS
#if __has_builtin(__builtin_amdgcn_s_wait_asynccnt)
  __builtin_amdgcn_s_wait_asynccnt(0);
#else
  asm volatile("s_wait_asynccnt 0x0" ::: "memory");
#endif
#endif
  __syncthreads();
}

// ---------------- kernel 1: router (one wave per token) ----------------
__global__ __launch_bounds__(256) void moe_router(
    const float* __restrict__ x, const float* __restrict__ rw,
    int* __restrict__ counts, int* __restrict__ tokidx, float* __restrict__ gateval) {
  __shared__ float srw[NEXP * HDIM];  // 32KB
  const int tid = threadIdx.x;
  for (int i = tid; i < NEXP * HDIM; i += 256) srw[i] = rw[i];
  __syncthreads();

  const int wave = tid >> 5, lane = tid & 31;
  const int t = blockIdx.x * 8 + wave;
  const float* xr = x + (size_t)t * HDIM;

  float acc[NEXP];
#pragma unroll
  for (int e = 0; e < NEXP; ++e) acc[e] = 0.f;
  for (int j = 0; j < HDIM / 32; ++j) {
    const int h = j * 32 + lane;
    const float xv = xr[h];
#pragma unroll
    for (int e = 0; e < NEXP; ++e) acc[e] += xv * srw[e * HDIM + h];
  }
#pragma unroll
  for (int e = 0; e < NEXP; ++e)
    for (int off = 16; off > 0; off >>= 1) acc[e] += __shfl_xor(acc[e], off, 32);

  if (lane == 0) {
    float m = acc[0];
    for (int e = 1; e < NEXP; ++e) m = fmaxf(m, acc[e]);
    float p[NEXP], s = 0.f;
    for (int e = 0; e < NEXP; ++e) { p[e] = expf(acc[e] - m); s += p[e]; }
    const float inv = 1.f / s;
    for (int e = 0; e < NEXP; ++e) p[e] *= inv;
    int i1 = 0;
    for (int e = 1; e < NEXP; ++e) if (p[e] > p[i1]) i1 = e;
    int i2 = (i1 == 0) ? 1 : 0;
    for (int e = 0; e < NEXP; ++e) if (e != i1 && p[e] > p[i2]) i2 = e;
    int pos = atomicAdd(&counts[i1], 1);
    if (pos < CAP) { tokidx[i1 * CAP + pos] = t; gateval[i1 * CAP + pos] = p[i1]; }
    pos = atomicAdd(&counts[i2], 1);
    if (pos < CAP) { tokidx[i2 * CAP + pos] = t; gateval[i2 * CAP + pos] = p[i2]; }
  }
}

// ---------------- kernel 2: x -> bf16 (packed, coalesced) ----------------
__global__ __launch_bounds__(256) void moe_cvt_x(const float* __restrict__ x,
                                                 unsigned* __restrict__ xbf) {
  const int i = blockIdx.x * 256 + threadIdx.x;  // over T*H/2 pairs (exact)
  const float2 v = reinterpret_cast<const float2*>(x)[i];
  xbf[i] = (unsigned)f2bf(v.x) | ((unsigned)f2bf(v.y) << 16);
}

// ---------------- kernel 2b: fp32 [R][C] -> bf16 transposed [C][R], per expert ----
__global__ __launch_bounds__(256) void moe_cvt_tr(const float* __restrict__ src,
                                                  ushort_t* __restrict__ dst,
                                                  int R, int C) {
  __shared__ float t[32][33];
  src += (size_t)blockIdx.z * R * C;
  dst += (size_t)blockIdx.z * R * C;
  const int c0 = blockIdx.x * 32, r0 = blockIdx.y * 32;
  const int tc = threadIdx.x & 31, tr = threadIdx.x >> 5;  // tr in 0..7
#pragma unroll
  for (int i = 0; i < 32; i += 8)
    t[tr + i][tc] = src[(size_t)(r0 + tr + i) * C + c0 + tc];
  __syncthreads();
#pragma unroll
  for (int i = 0; i < 32; i += 8)
    dst[(size_t)(c0 + tr + i) * R + r0 + tc] = f2bf(t[tc][tr + i]);
}

// ---------------- kernel 3: exclusive prefix over expert counts ----------------
__global__ void moe_offsets(const int* __restrict__ counts, int* __restrict__ offsets) {
  if (threadIdx.x == 0) {
    int s = 0;
    for (int e = 0; e < NEXP; ++e) {
      offsets[e] = s;
      int c = counts[e]; if (c > CAP) c = CAP;
      s += c;
    }
    offsets[NEXP] = s;
  }
}

// ---------------- kernel 4: grouped GEMM1 + bias + GELU -> bf16 hid ----------------
// block tile 128(M) x 256(N); 8 waves of 64x64; K staged 32 at a time.
__global__ __launch_bounds__(256) void moe_gemm1(
    const ushort_t* __restrict__ xbf, const ushort_t* __restrict__ w1t,
    const float* __restrict__ b1, const int* __restrict__ counts,
    const int* __restrict__ offsets, const int* __restrict__ tokidx,
    ushort_t* __restrict__ hbf) {
  __shared__ ushort_t lA[128 * SA];
  __shared__ ushort_t lB[256 * SA];
  __shared__ int   sTok[128];
  __shared__ float sBias[256];

  const int e = blockIdx.x >> 5;          // CAP/128 = 32 M-tiles per expert
  const int mt = blockIdx.x & 31;
  int cnt = counts[e]; if (cnt > CAP) cnt = CAP;
  const int row0 = mt * 128;
  if (row0 >= cnt) return;
  const int nBase = blockIdx.y * 256;
  const int gBase = offsets[e] + row0;
  const int tid = threadIdx.x;

  if (tid < 128) {
    const int r = row0 + tid;
    sTok[tid] = (r < cnt) ? tokidx[e * CAP + r] : 0;
  }
  sBias[tid] = b1[(size_t)e * FDIM + nBase + tid];
  __syncthreads();  // sTok ready for staging

  // staging addresses: A = 128 gathered bf16 token rows; B = 256 bf16 w1t rows
  const int am = tid >> 1, ac = tid & 1;
  const ushort_t* aSrc = xbf + (size_t)sTok[am] * HDIM + ac * 16;
  ushort_t*       aDst = lA + am * SA + ac * 16;
  const ushort_t* bSrc = w1t + ((size_t)e * FDIM + nBase + tid) * HDIM;
  ushort_t*       bDst = lB + tid * SA;

  const int lane = tid & 31, wave = tid >> 5;
  const int wm = wave & 1, wn = wave >> 1;   // 2x4 wave grid: 64x64 per wave
  const int half = lane >> 4, l16 = lane & 15;

  const v8f vz = {0.f, 0.f, 0.f, 0.f, 0.f, 0.f, 0.f, 0.f};
  v8f acc[4][4];
#pragma unroll
  for (int mi = 0; mi < 4; ++mi)
#pragma unroll
    for (int ni = 0; ni < 4; ++ni) acc[mi][ni] = vz;

  for (int k0 = 0; k0 < HDIM; k0 += 32) {
    copy16(aSrc + k0, aDst);
    copy16(aSrc + k0 + 8, aDst + 8);
#pragma unroll
    for (int j = 0; j < 4; ++j) copy16(bSrc + k0 + j * 8, bDst + j * 8);
    lds_stage_fence();

    v16bf a[4], b[4];
#pragma unroll
    for (int mi = 0; mi < 4; ++mi) a[mi] = load_fragA(lA, wm * 64 + mi * 16 + l16, half);
#pragma unroll
    for (int ni = 0; ni < 4; ++ni) b[ni] = load_fragB(lB, wn * 64 + ni * 16 + l16, half);
#pragma unroll
    for (int mi = 0; mi < 4; ++mi)
#pragma unroll
      for (int ni = 0; ni < 4; ++ni)
        acc[mi][ni] = __builtin_amdgcn_wmma_f32_16x16x32_bf16(
            false, a[mi], false, b[ni], (short)0, acc[mi][ni], false, false);
    __syncthreads();
  }

  // epilogue: bias + exact GELU -> bf16 intermediate
#pragma unroll
  for (int mi = 0; mi < 4; ++mi)
#pragma unroll
    for (int ni = 0; ni < 4; ++ni)
#pragma unroll
      for (int i = 0; i < 8; ++i) {
        const int mLoc = wm * 64 + mi * 16 + half * 8 + i;
        if (row0 + mLoc < cnt) {
          const int nLoc = wn * 64 + ni * 16 + l16;
          float v = acc[mi][ni][i] + sBias[nLoc];
          v = 0.5f * v * (1.0f + erff(v * 0.70710678118654752f));
          hbf[(size_t)(gBase + mLoc) * FDIM + nBase + nLoc] = f2bf(v);
        }
      }
}

// ---------------- kernel 5: grouped GEMM2 + bias + gate*scatter-add ----------------
__global__ __launch_bounds__(256) void moe_gemm2(
    const ushort_t* __restrict__ hbf, const ushort_t* __restrict__ w2t,
    const float* __restrict__ b2, const int* __restrict__ counts,
    const int* __restrict__ offsets, const int* __restrict__ tokidx,
    const float* __restrict__ gateval, float* __restrict__ out) {
  __shared__ ushort_t lA[128 * SA];
  __shared__ ushort_t lB[256 * SA];
  __shared__ int   sTok[128];
  __shared__ float sGate[128];
  __shared__ float sBias[256];

  const int e = blockIdx.x >> 5;
  const int mt = blockIdx.x & 31;
  int cnt = counts[e]; if (cnt > CAP) cnt = CAP;
  const int row0 = mt * 128;
  if (row0 >= cnt) return;
  const int nBase = blockIdx.y * 256;
  const int gBase = offsets[e] + row0;
  const int tid = threadIdx.x;

  if (tid < 128) {
    const int r = row0 + tid;
    sTok[tid]  = (r < cnt) ? tokidx[e * CAP + r] : 0;
    sGate[tid] = (r < cnt) ? gateval[e * CAP + r] : 0.f;
  }
  sBias[tid] = b2[(size_t)e * HDIM + nBase + tid];
  __syncthreads();

  const int am = tid >> 1, ac = tid & 1;
  const ushort_t* aSrc = hbf + (size_t)(gBase + am) * FDIM + ac * 16;
  ushort_t*       aDst = lA + am * SA + ac * 16;
  const ushort_t* bSrc = w2t + ((size_t)e * HDIM + nBase + tid) * FDIM;
  ushort_t*       bDst = lB + tid * SA;

  const int lane = tid & 31, wave = tid >> 5;
  const int wm = wave & 1, wn = wave >> 1;
  const int half = lane >> 4, l16 = lane & 15;

  const v8f vz = {0.f, 0.f, 0.f, 0.f, 0.f, 0.f, 0.f, 0.f};
  v8f acc[4][4];
#pragma unroll
  for (int mi = 0; mi < 4; ++mi)
#pragma unroll
    for (int ni = 0; ni < 4; ++ni) acc[mi][ni] = vz;

  for (int k0 = 0; k0 < FDIM; k0 += 32) {
    copy16(aSrc + k0, aDst);
    copy16(aSrc + k0 + 8, aDst + 8);
#pragma unroll
    for (int j = 0; j < 4; ++j) copy16(bSrc + k0 + j * 8, bDst + j * 8);
    lds_stage_fence();

    v16bf a[4], b[4];
#pragma unroll
    for (int mi = 0; mi < 4; ++mi) a[mi] = load_fragA(lA, wm * 64 + mi * 16 + l16, half);
#pragma unroll
    for (int ni = 0; ni < 4; ++ni) b[ni] = load_fragB(lB, wn * 64 + ni * 16 + l16, half);
#pragma unroll
    for (int mi = 0; mi < 4; ++mi)
#pragma unroll
      for (int ni = 0; ni < 4; ++ni)
        acc[mi][ni] = __builtin_amdgcn_wmma_f32_16x16x32_bf16(
            false, a[mi], false, b[ni], (short)0, acc[mi][ni], false, false);
    __syncthreads();
  }

  // (acc + b2) * gate, scatter-add (exactly 2 float adds per out element -> deterministic)
#pragma unroll
  for (int mi = 0; mi < 4; ++mi)
#pragma unroll
    for (int ni = 0; ni < 4; ++ni)
#pragma unroll
      for (int i = 0; i < 8; ++i) {
        const int mLoc = wm * 64 + mi * 16 + half * 8 + i;
        if (row0 + mLoc < cnt) {
          const int nLoc = wn * 64 + ni * 16 + l16;
          const float v = (acc[mi][ni][i] + sBias[nLoc]) * sGate[mLoc];
          const int t = sTok[mLoc];
          unsafeAtomicAdd(&out[(size_t)t * HDIM + nBase + nLoc], v);
        }
      }
}

// ---------------- host launcher ----------------
extern "C" void kernel_launch(void* const* d_in, const int* in_sizes, int n_in,
                              void* d_out, int out_size, void* d_ws, size_t ws_size,
                              hipStream_t stream) {
  (void)in_sizes; (void)n_in; (void)out_size; (void)ws_size;
  const float* x  = (const float*)d_in[0];
  const float* rw = (const float*)d_in[1];
  const float* w1 = (const float*)d_in[2];
  const float* b1 = (const float*)d_in[3];
  const float* w2 = (const float*)d_in[4];
  const float* b2 = (const float*)d_in[5];
  float* out = (float*)d_out;

  // ws carve-up (256B aligned): counts, offsets, token lists, gates,
  // bf16 x (16MB), bf16 hid (~136MB), bf16 transposed w1/w2 (67MB each)
  char* ws = (char*)d_ws;
  size_t cur = 0;
  auto carve = [&](size_t bytes) -> char* {
    char* p = ws + cur;
    cur = (cur + bytes + 255) & ~(size_t)255;
    return p;
  };
  int*      counts  = (int*)carve(NEXP * sizeof(int));
  int*      offsets = (int*)carve((NEXP + 1) * sizeof(int));
  int*      tokidx  = (int*)carve((size_t)NEXP * CAP * sizeof(int));
  float*    gatevl  = (float*)carve((size_t)NEXP * CAP * sizeof(float));
  ushort_t* xbf     = (ushort_t*)carve((size_t)T_TOK * HDIM * sizeof(ushort_t));
  ushort_t* hbf     = (ushort_t*)carve((size_t)(2 * T_TOK + 256) * FDIM * sizeof(ushort_t));
  ushort_t* w1t     = (ushort_t*)carve((size_t)NEXP * HDIM * FDIM * sizeof(ushort_t));
  ushort_t* w2t     = (ushort_t*)carve((size_t)NEXP * FDIM * HDIM * sizeof(ushort_t));

  (void)hipMemsetAsync(counts, 0, NEXP * sizeof(int), stream);
  (void)hipMemsetAsync(out, 0, (size_t)T_TOK * HDIM * sizeof(float), stream);

  moe_router<<<T_TOK / 8, 256, 0, stream>>>(x, rw, counts, tokidx, gatevl);
  moe_cvt_x<<<(T_TOK * HDIM / 2) / 256, 256, 0, stream>>>(x, (unsigned*)xbf);
  // w1 [E][H][F] -> w1t [E][F][H];  w2 [E][F][H] -> w2t [E][H][F]
  moe_cvt_tr<<<dim3(FDIM / 32, HDIM / 32, NEXP), 256, 0, stream>>>(w1, w1t, HDIM, FDIM);
  moe_cvt_tr<<<dim3(HDIM / 32, FDIM / 32, NEXP), 256, 0, stream>>>(w2, w2t, FDIM, HDIM);
  moe_offsets<<<1, 32, 0, stream>>>(counts, offsets);
  moe_gemm1<<<dim3(NEXP * (CAP / 128), FDIM / 256), 256, 0, stream>>>(
      xbf, w1t, b1, counts, offsets, tokidx, hbf);
  moe_gemm2<<<dim3(NEXP * (CAP / 128), HDIM / 256), 256, 0, stream>>>(
      hbf, w2t, b2, counts, offsets, tokidx, gatevl, out);
}
